// TS_SUB_89146341196410
// MI455X (gfx1250) — matrix operand order repaired and checked by reading, not compiled, not run on hardware
//
#include <hip/hip_runtime.h>
#include <hip/hip_bf16.h>
#include <math.h>

typedef __attribute__((ext_vector_type(16))) __bf16 v16bf;
typedef __attribute__((ext_vector_type(8)))  __bf16 v8bf;
typedef __attribute__((ext_vector_type(8)))  float  v8f;

#define GEMM_NT 4   // 4 column tiles per wave
#define GEMM_MT 2   // 2 row tiles per wave -> 32x64 strip, B fragments reused across M

// ---------------------------------------------------------------------------
// Utility kernels
// ---------------------------------------------------------------------------
__global__ void fill_f32_kernel(float* p, float v, long n) {
    long i = (long)blockIdx.x * blockDim.x + threadIdx.x;
    if (i < n) p[i] = v;
}

__global__ void cast_bf16_kernel(const float* __restrict__ x, __bf16* __restrict__ y, long n) {
    long i = (long)blockIdx.x * blockDim.x + threadIdx.x;
    if (i < n) y[i] = (__bf16)x[i];
}

// W [K,N] f32 (row-major) -> Wt [N,K] bf16 (row-major) so GEMM B-lanes read contiguous K runs
__global__ void transpose_cast_kernel(const float* __restrict__ W, __bf16* __restrict__ Wt,
                                      int K, int N) {
    long i = (long)blockIdx.x * blockDim.x + threadIdx.x;
    if (i >= (long)K * N) return;
    int k = (int)(i / N), n = (int)(i % N);
    Wt[(size_t)n * K + k] = (__bf16)W[i];
}

// ---------------------------------------------------------------------------
// WMMA GEMM:  C[M,N] = A[M,K](bf16) * Bt[N,K]^T(bf16) (+bias) (+relu)
// One wave (32 threads) per block, computes a 32x64 strip (MT=2 x NT=4 tiles).
// Each B fragment feeds 2 WMMAs (both M tiles) -> 12 VMEM : 8 WMMA per k-step.
// A-fragment (16-bit A 16x32), contiguous K order like B (not the ISA table order):
//   lane half h = lane>>4, row m = lane&15
//   elements 0..7  <-> K = 16h + e          (VGPRs 0..3)
//   elements 8..15 <-> K = 16h + e (VGPRs 4..7)
// B-fragment (32x16): col n = lane&15, element e <-> K = 16h + e (contiguous in Bt row)
// C/D f32 16x16: col = lane&15, rows r+8h for accumulator element r.
// ---------------------------------------------------------------------------
__global__ void gemm_bf16_wmma_kernel(const __bf16* __restrict__ A,
                                      const __bf16* __restrict__ Bt,
                                      const float* __restrict__ bias,
                                      float* __restrict__ Cf,
                                      __bf16* __restrict__ Cb,
                                      int M, int N, int K, int relu) {
    const int lane = threadIdx.x & 31;
    const int half = lane >> 4;
    const int lm   = lane & 15;
    const int tm   = blockIdx.y;                  // 32-row strip index
    const int cn0  = blockIdx.x * (GEMM_NT * 16);

    const __bf16* arow[GEMM_MT];
#pragma unroll
    for (int m = 0; m < GEMM_MT; ++m)
        arow[m] = A + (size_t)(tm * (16 * GEMM_MT) + m * 16 + lm) * K;
    const __bf16* brow[GEMM_NT];
#pragma unroll
    for (int t = 0; t < GEMM_NT; ++t)
        brow[t] = Bt + (size_t)(cn0 + t * 16 + lm) * K;

    v8f acc[GEMM_MT][GEMM_NT];
#pragma unroll
    for (int m = 0; m < GEMM_MT; ++m)
#pragma unroll
        for (int t = 0; t < GEMM_NT; ++t)
#pragma unroll
            for (int i = 0; i < 8; ++i) acc[m][t][i] = 0.0f;

    for (int k0 = 0; k0 < K; k0 += 32) {
        v16bf a[GEMM_MT];
#pragma unroll
        for (int m = 0; m < GEMM_MT; ++m) {
            a[m] = *(const v16bf*)(arow[m] + k0 + 16 * half);          // K = 16h .. 16h+15
            if (k0 + 32 < K) __builtin_prefetch(arow[m] + k0 + 64, 0, 1); // global_prefetch_b8
        }
#pragma unroll
        for (int t = 0; t < GEMM_NT; ++t) {
            v16bf b = *(const v16bf*)(brow[t] + k0 + 16 * half);        // K = 16h .. 16h+15
#pragma unroll
            for (int m = 0; m < GEMM_MT; ++m)
                acc[m][t] = __builtin_amdgcn_wmma_f32_16x16x32_bf16(
                    false, a[m], false, b, (short)0, acc[m][t], false, false);
        }
    }

#pragma unroll
    for (int m = 0; m < GEMM_MT; ++m)
#pragma unroll
        for (int t = 0; t < GEMM_NT; ++t) {
            int col = cn0 + t * 16 + lm;
            float bv = bias ? bias[col] : 0.0f;
#pragma unroll
            for (int r = 0; r < 8; ++r) {
                int row = tm * (16 * GEMM_MT) + m * 16 + r + 8 * half;
                float v = acc[m][t][r] + bv;
                if (relu) v = fmaxf(v, 0.0f);
                size_t idx = (size_t)row * N + col;
                if (Cf) Cf[idx] = v;
                if (Cb) Cb[idx] = (__bf16)v;
            }
        }
}

// ---------------------------------------------------------------------------
// GCN kernels
// ---------------------------------------------------------------------------
__global__ void degree_kernel(const int* __restrict__ src, const int* __restrict__ dst,
                              float* dgo, float* dgi, int E) {
    int i = blockIdx.x * blockDim.x + threadIdx.x;
    if (i >= E) return;
    atomicAdd(&dgo[src[i]], 1.0f);
    atomicAdd(&dgi[dst[i]], 1.0f);
}

__global__ void rsqrt_clamp_kernel(float* d, int n) {
    int i = blockIdx.x * blockDim.x + threadIdx.x;
    if (i < n) d[i] = rsqrtf(fmaxf(d[i], 1.0f));
}

// agg[dst] += h[src] * cs[src]   (block per edge, thread per feature)
__global__ void gcn_agg_kernel(const float* __restrict__ h, const int* __restrict__ src,
                               const int* __restrict__ dst, const float* __restrict__ cs,
                               float* agg, int F) {
    int e = blockIdx.x, f = threadIdx.x;
    int s = src[e], d = dst[e];
    atomicAdd(&agg[(size_t)d * F + f], h[(size_t)s * F + f] * cs[s]);
}

// res (in: relu(x@rW+rb)) -> h2 = relu(agg*cd) + res, in place
__global__ void gcn_finalize_kernel(float* res, const float* __restrict__ agg,
                                    const float* __restrict__ cd, long n, int cols) {
    long i = (long)blockIdx.x * blockDim.x + threadIdx.x;
    if (i >= n) return;
    int r = (int)(i / cols);
    res[i] = fmaxf(agg[i] * cd[r], 0.0f) + res[i];
}

// per-column mean + rstd over `rows`
__global__ void col_stats_kernel(const float* __restrict__ x, int rows, int cols,
                                 float* mu, float* rstd) {
    __shared__ float s1[256], s2[256];
    int c = blockIdx.x;
    float a = 0.0f, b = 0.0f;
    for (int r = threadIdx.x; r < rows; r += blockDim.x) {
        float v = x[(size_t)r * cols + c];
        a += v; b += v * v;
    }
    s1[threadIdx.x] = a; s2[threadIdx.x] = b;
    __syncthreads();
    for (int s = 128; s; s >>= 1) {
        if ((int)threadIdx.x < s) { s1[threadIdx.x] += s1[threadIdx.x + s];
                                    s2[threadIdx.x] += s2[threadIdx.x + s]; }
        __syncthreads();
    }
    if (threadIdx.x == 0) {
        float m = s1[0] / rows;
        float var = s2[0] / rows - m * m;
        mu[c] = m;
        rstd[c] = rsqrtf(var + 1e-5f);
    }
}

__global__ void bn_apply_kernel(const float* __restrict__ x, const float* __restrict__ mu,
                                const float* __restrict__ rstd, const float* __restrict__ g,
                                const float* __restrict__ be, float* yf, __bf16* yb,
                                long n, int cols, int relu) {
    long i = (long)blockIdx.x * blockDim.x + threadIdx.x;
    if (i >= n) return;
    int c = (int)(i % cols);
    float v = g[c] * (x[i] - mu[c]) * rstd[c] + be[c];
    if (relu) v = fmaxf(v, 0.0f);
    if (yf) yf[i] = v;
    if (yb) yb[i] = (__bf16)v;
}

// ---------------------------------------------------------------------------
// GAT kernels (H=4 heads, C=128 per head, 512 feats)
// ---------------------------------------------------------------------------
// al[node][h] = sum_c hs[node][h*128+c] * a[h*128+c]; one wave per head (wave32)
__global__ void gat_al_kernel(const float* __restrict__ hs, const float* __restrict__ a,
                              float* al, int nNodes) {
    int node = blockIdx.x;
    int w = threadIdx.x >> 5, lane = threadIdx.x & 31;
    const float* row = hs + (size_t)node * 512 + w * 128;
    const float* av  = a + w * 128;
    float s = 0.0f;
#pragma unroll
    for (int c = lane; c < 128; c += 32) s += row[c] * av[c];
    for (int o = 16; o; o >>= 1) s += __shfl_down(s, o, 32);
    if (lane == 0) al[node * 4 + w] = s;
}

__device__ inline void atomicMaxF32(float* addr, float val) {
    if (val >= 0.0f) atomicMax((int*)addr, __float_as_int(val));
    else             atomicMin((unsigned int*)addr, __float_as_uint(val));
}

__global__ void gat_logits_kernel(const float* __restrict__ al_s, const float* __restrict__ al_d,
                                  const int* __restrict__ src, const int* __restrict__ dst,
                                  float* eL, float* mx, int E) {
    int i = blockIdx.x * blockDim.x + threadIdx.x;
    if (i >= E * 4) return;
    int e = i >> 2, h = i & 3;
    float v = al_s[src[e] * 4 + h] + al_d[dst[e] * 4 + h];
    v = (v > 0.0f) ? v : 0.2f * v;       // leaky_relu 0.2
    eL[i] = v;
    atomicMaxF32(&mx[dst[e] * 4 + h], v);
}

__global__ void gat_exp_kernel(float* eL, const float* __restrict__ mx, float* denom,
                               const int* __restrict__ dst, int E) {
    int i = blockIdx.x * blockDim.x + threadIdx.x;
    if (i >= E * 4) return;
    int e = i >> 2, h = i & 3;
    int d = dst[e];
    float ev = __expf(eL[i] - mx[d * 4 + h]);
    eL[i] = ev;
    atomicAdd(&denom[d * 4 + h], ev);
}

// out[dst] += hs[src] * alpha ; block per edge, 128 threads x float4
__global__ void gat_aggregate_kernel(const float* __restrict__ hs, const float* __restrict__ ee,
                                     const float* __restrict__ denom, const int* __restrict__ src,
                                     const int* __restrict__ dst, float* outp, int E) {
    int e = blockIdx.x, t = threadIdx.x;
    int s = src[e], d = dst[e];
    int h = t >> 5;                                       // feature 4t -> head (4t)/128
    float a = ee[e * 4 + h] / (denom[d * 4 + h] + 1e-16f);
    float4 v = ((const float4*)(hs + (size_t)s * 512))[t];
    float* orow = outp + (size_t)d * 512 + t * 4;
    atomicAdd(orow + 0, v.x * a);
    atomicAdd(orow + 1, v.y * a);
    atomicAdd(orow + 2, v.z * a);
    atomicAdd(orow + 3, v.w * a);
}

// dst[batch[node]*stride + t] += rep[node][t] + bias[t]
__global__ void batch_sum_kernel(const float* __restrict__ rep, const float* __restrict__ bias,
                                 const int* __restrict__ batch, float* dst, int stride, int nNodes) {
    int node = blockIdx.x, t = threadIdx.x;
    float v = rep[(size_t)node * 512 + t] + bias[t];
    atomicAdd(&dst[(size_t)batch[node] * stride + t], v);
}

// ---------------------------------------------------------------------------
// Head kernels
// ---------------------------------------------------------------------------
__global__ void pf_kernel(const float* __restrict__ p, const float* __restrict__ W,
                          const float* __restrict__ b, float* out, int B, int D) {
    int i = blockIdx.x * blockDim.x + threadIdx.x;
    if (i >= B * D) return;
    int bb = i / D, j = i % D;
    out[i] = p[bb] * W[j] + b[j];
}

__global__ void concat_cast_kernel(const float* __restrict__ s1, const float* __restrict__ s2,
                                   __bf16* drug, int B) {
    int i = blockIdx.x * blockDim.x + threadIdx.x;
    if (i >= B * 2048) return;
    int b = i / 2048, j = i % 2048;
    float v = (j < 1024) ? s1[(size_t)b * 1024 + j] : s2[(size_t)b * 1024 + j - 1024];
    drug[i] = (__bf16)v;
}

__global__ void out_head_kernel(const float* __restrict__ h, const float* __restrict__ Wo,
                                const float* __restrict__ bo, float* out, int K) {
    int b = blockIdx.x, lane = threadIdx.x;
    const float* row = h + (size_t)b * K;
    float s = 0.0f;
    for (int k = lane; k < K; k += 32) s += row[k] * Wo[k];
    for (int o = 16; o; o >>= 1) s += __shfl_down(s, o, 32);
    if (lane == 0) out[b] = 1.0f / (1.0f + __expf(-(s + bo[0])));
}

// ---------------------------------------------------------------------------
// Host orchestration
// ---------------------------------------------------------------------------
extern "C" void kernel_launch(void* const* d_in, const int* in_sizes, int n_in,
                              void* d_out, int out_size, void* d_ws, size_t ws_size,
                              hipStream_t stream) {
    (void)in_sizes; (void)n_in; (void)ws_size;
    const int NN = 20480, EE = 163840, EB = 163840, BB = 512;

    // ---- inputs ----
    const float* x_core = (const float*)d_in[0];
    const float* x_sub1 = (const float*)d_in[1];
    const float* x_sub2 = (const float*)d_in[2];
    const int* e_core = (const int*)d_in[3];
    const int* e_sub1 = (const int*)d_in[4];
    const int* e_sub2 = (const int*)d_in[5];
    const int* b1e = (const int*)d_in[6];
    const int* b2e = (const int*)d_in[7];
    const int* core_batch = (const int*)d_in[8];
    const int* sub1_batch = (const int*)d_in[9];
    const int* sub2_batch = (const int*)d_in[10];
    const float* p1 = (const float*)d_in[11];
    const float* p2 = (const float*)d_in[12];
    // Param leaves (JAX pytree order: dict keys sorted alphabetically):
    // 13..16 fc1(W,b,g,be)  17..20 fc2  21..25 gat(Wsrc,Wdst,a_src,a_dst,bias)
    // 26..31 gcn_core.l1(W,b,rW,rb,g,be) 32..37 .l2  38..39 .proj(W,b)
    // 40..45 gcn_sub.l1 46..51 .l2 52..53 .proj  54..55 out(W,b)  56..57 pf1  58..59 pf2
    auto P = [&](int i) { return (const float*)d_in[i]; };

    // ---- workspace arena ----
    size_t off = 0;
    auto alloc = [&](size_t bytes) -> void* {
        void* p = (char*)d_ws + off;
        off += (bytes + 255) & ~(size_t)255;
        return p;
    };
    __bf16* enc[3];
    for (int g = 0; g < 3; ++g) enc[g] = (__bf16*)alloc((size_t)NN * 256 * 2);
    __bf16* xb   = (__bf16*)alloc((size_t)NN * 64 * 2);
    float* hbuf  = (float*)alloc((size_t)NN * 64 * 4);
    float* resb  = (float*)alloc((size_t)NN * 64 * 4);
    float* aggb  = (float*)alloc((size_t)NN * 64 * 4);
    float* cs    = (float*)alloc((size_t)NN * 4);
    float* cd    = (float*)alloc((size_t)NN * 4);
    float* mu    = (float*)alloc(1024 * 4);
    float* rstd  = (float*)alloc(1024 * 4);
    __bf16* wt   = (__bf16*)alloc((size_t)2048 * 1024 * 2);   // max weight (fc1)
    float* buf0  = (float*)alloc((size_t)NN * 512 * 4);       // hs
    float* buf1  = (float*)alloc((size_t)NN * 512 * 4);       // hd
    float* buf2  = (float*)alloc((size_t)NN * 512 * 4);       // rep
    float* al_s  = (float*)alloc((size_t)NN * 4 * 4);
    float* al_d  = (float*)alloc((size_t)NN * 4 * 4);
    float* eL    = (float*)alloc((size_t)EB * 4 * 4);
    float* mx    = (float*)alloc((size_t)NN * 4 * 4);
    float* denom = (float*)alloc((size_t)NN * 4 * 4);
    __bf16* drug = (__bf16*)alloc((size_t)BB * 2048 * 2);
    float* fcA   = (float*)alloc((size_t)BB * 1024 * 4);
    __bf16* fcAb = (__bf16*)alloc((size_t)BB * 1024 * 2);
    float* fcB   = (float*)alloc((size_t)BB * 512 * 4);

    auto fill = [&](float* p, float v, long n) {
        fill_f32_kernel<<<(unsigned)((n + 255) / 256), 256, 0, stream>>>(p, v, n);
    };
    // C = A(bf16,[M,K]) @ W(f32,[K,N]) (+bias)(+relu) -> Cf/Cb; transposes W to bf16 first
    auto gemm = [&](const __bf16* A, const float* W, const float* bias,
                    float* Cf, __bf16* Cb, int M, int N, int K, int relu) {
        long wn = (long)K * N;
        transpose_cast_kernel<<<(unsigned)((wn + 255) / 256), 256, 0, stream>>>(W, wt, K, N);
        dim3 grid(N / (GEMM_NT * 16), M / (GEMM_MT * 16));
        gemm_bf16_wmma_kernel<<<grid, 32, 0, stream>>>(A, wt, bias, Cf, Cb, M, N, K, relu);
    };

    // ---- zero outputs (s1/s2 are atomically accumulated) ----
    fill((float*)d_out, 0.0f, out_size);
    const size_t SEG = (size_t)BB * 1024;
    float* outv = (float*)d_out;
    float* s1r  = outv + 512;
    float* p1fr = outv + 512 + SEG;
    float* s2r  = outv + 512 + 2 * SEG;
    float* p2fr = outv + 512 + 3 * SEG;

    // ---- GCN encoder per graph ----
    auto run_gcn = [&](const float* x, const int* ei, int pb, __bf16* enc_out) {
        const int* src = ei; const int* dst = ei + EE;
        fill(cs, 0.0f, NN); fill(cd, 0.0f, NN);
        degree_kernel<<<(EE + 255) / 256, 256, 0, stream>>>(src, dst, cs, cd, EE);
        rsqrt_clamp_kernel<<<(NN + 255) / 256, 256, 0, stream>>>(cs, NN);
        rsqrt_clamp_kernel<<<(NN + 255) / 256, 256, 0, stream>>>(cd, NN);
        cast_bf16_kernel<<<(unsigned)(((long)NN * 32 + 255) / 256), 256, 0, stream>>>(x, xb, (long)NN * 32);
        for (int layer = 0; layer < 2; ++layer) {
            int din = layer ? 64 : 32;
            int q = pb + layer * 6;
            gemm(xb, P(q + 0), P(q + 1), hbuf, nullptr, NN, 64, din, 0);  // h = x@W+b
            gemm(xb, P(q + 2), P(q + 3), resb, nullptr, NN, 64, din, 1);  // relu(x@rW+rb)
            fill(aggb, 0.0f, (long)NN * 64);
            gcn_agg_kernel<<<EE, 64, 0, stream>>>(hbuf, src, dst, cs, aggb, 64);
            gcn_finalize_kernel<<<(unsigned)(((long)NN * 64 + 255) / 256), 256, 0, stream>>>(
                resb, aggb, cd, (long)NN * 64, 64);
            col_stats_kernel<<<64, 256, 0, stream>>>(resb, NN, 64, mu, rstd);
            bn_apply_kernel<<<(unsigned)(((long)NN * 64 + 255) / 256), 256, 0, stream>>>(
                resb, mu, rstd, P(q + 4), P(q + 5), nullptr, xb, (long)NN * 64, 64, 0);
        }
        // proj (+relu fused: encodings are only consumed through relu in _inter_att)
        gemm(xb, P(pb + 12), P(pb + 13), nullptr, enc_out, NN, 256, 64, 1);
    };
    run_gcn(x_core, e_core, 26, enc[0]);
    run_gcn(x_sub1, e_sub1, 40, enc[1]);
    run_gcn(x_sub2, e_sub2, 40, enc[2]);

    // ---- GAT inter-attention ----
    const float* Wsrc = P(21); const float* Wdst = P(22);
    const float* aS = P(23);   const float* aD = P(24);   const float* gbias = P(25);
    auto gat_dir = [&](const __bf16* xsrc, const __bf16* xdst, const int* src, const int* dst,
                       const int* batch, float* sdst /* [B,1024] region, col offset applied */) {
        gemm(xsrc, Wsrc, nullptr, buf0, nullptr, NN, 512, 256, 0);   // hs
        gemm(xdst, Wdst, nullptr, buf1, nullptr, NN, 512, 256, 0);   // hd
        gat_al_kernel<<<NN, 128, 0, stream>>>(buf0, aS, al_s, NN);
        gat_al_kernel<<<NN, 128, 0, stream>>>(buf1, aD, al_d, NN);
        fill(mx, -1e30f, (long)NN * 4);
        fill(denom, 0.0f, (long)NN * 4);
        gat_logits_kernel<<<(EB * 4 + 255) / 256, 256, 0, stream>>>(al_s, al_d, src, dst, eL, mx, EB);
        gat_exp_kernel<<<(EB * 4 + 255) / 256, 256, 0, stream>>>(eL, mx, denom, dst, EB);
        fill(buf2, 0.0f, (long)NN * 512);
        gat_aggregate_kernel<<<EB, 128, 0, stream>>>(buf0, eL, denom, src, dst, buf2, EB);
        batch_sum_kernel<<<NN, 512, 0, stream>>>(buf2, gbias, batch, sdst, 1024, NN);
    };
    auto inter_att = [&](const __bf16* hr, const __bf16* tr, const int* ei,
                         const int* hbatch, const int* tbatch, float* sreg) {
        const int* es = ei; const int* ed = ei + EB;
        gat_dir(hr, tr, es, ed, tbatch, sreg + 512); // t_rep -> cols [512,1024)
        gat_dir(tr, hr, ed, es, hbatch, sreg + 0);   // h_rep -> cols [0,512)
    };
    inter_att(enc[0], enc[1], b1e, core_batch, sub1_batch, s1r);
    inter_att(enc[0], enc[2], b2e, core_batch, sub2_batch, s2r);

    // ---- p features ----
    pf_kernel<<<(BB * 1024 + 255) / 256, 256, 0, stream>>>(p1, P(56), P(57), p1fr, BB, 1024);
    pf_kernel<<<(BB * 1024 + 255) / 256, 256, 0, stream>>>(p2, P(58), P(59), p2fr, BB, 1024);

    // ---- MLP head: drug = [s1|s2] -> fc1+bn+relu -> fc2+bn+relu -> sigmoid ----
    concat_cast_kernel<<<(BB * 2048 + 255) / 256, 256, 0, stream>>>(s1r, s2r, drug, BB);
    gemm(drug, P(13), P(14), fcA, nullptr, BB, 1024, 2048, 0);
    col_stats_kernel<<<1024, 256, 0, stream>>>(fcA, BB, 1024, mu, rstd);
    bn_apply_kernel<<<(unsigned)(((long)BB * 1024 + 255) / 256), 256, 0, stream>>>(
        fcA, mu, rstd, P(15), P(16), nullptr, fcAb, (long)BB * 1024, 1024, 1);
    gemm(fcAb, P(17), P(18), fcB, nullptr, BB, 512, 1024, 0);
    col_stats_kernel<<<512, 256, 0, stream>>>(fcB, BB, 512, mu, rstd);
    bn_apply_kernel<<<(unsigned)(((long)BB * 512 + 255) / 256), 256, 0, stream>>>(
        fcB, mu, rstd, P(19), P(20), fcB, nullptr, (long)BB * 512, 512, 1);
    out_head_kernel<<<BB, 32, 0, stream>>>(fcB, P(54), P(55), outv, 512);
}
